// Critic_17746804867537
// MI455X (gfx1250) — matrix operand order repaired and checked by reading, not compiled, not run on hardware
//
#include <hip/hip_runtime.h>
#include <hip/hip_bf16.h>

#define NUM_ITEMS 100000
#define D 128
#define H 256
#define Bb 128
#define S 200
#define Kslate 10
#define C 1000
#define IN_DIM 768
#define BS (Bb * S)   // 25600
#define BC (Bb * C)   // 128000
#define ALPHA 0.1f

typedef __attribute__((ext_vector_type(16))) __bf16 bf16x16;
typedef __attribute__((ext_vector_type(8)))  float  f32x8;
typedef __attribute__((ext_vector_type(4)))  int    v4i;
typedef v4i __attribute__((address_space(1)))* v4i_gp;   // global
typedef v4i __attribute__((address_space(3)))* v4i_lp;   // LDS

#if __has_builtin(__builtin_amdgcn_global_load_async_to_lds_b128) && \
    __has_builtin(__builtin_amdgcn_s_wait_asynccnt)
#define HAVE_ASYNC_LDS 1
#else
#define HAVE_ASYNC_LDS 0
#endif

union ABfrag { bf16x16 v; unsigned short u[16]; };

__device__ __forceinline__ unsigned short f2bf(float f) {
  union { float f; unsigned u; } x; x.f = f;
  unsigned r = x.u + 0x7FFFu + ((x.u >> 16) & 1u);   // round-to-nearest-even
  return (unsigned short)(r >> 16);
}
__device__ __forceinline__ float bf2f(unsigned short h) {
  union { unsigned u; float f; } x; x.u = ((unsigned)h) << 16; return x.f;
}
__device__ __forceinline__ float sigmf(float x) { return 1.0f / (1.0f + __expf(-x)); }

// A fragment 16x32 bf16 (ISA 7.12.2): lanes 0-15 row M=lane&15, elems 0..7 -> K+0..7,
// elems 8..15 -> K+16..23; lanes 16-31 add +8 to K. rstride allows strided rows.
__device__ __forceinline__ bf16x16 load_a(const unsigned short* p, int ld, int row0,
                                          int rstride, int k0, int lane) {
  ABfrag a;
  const unsigned short* r = p + (size_t)(row0 + (lane & 15) * rstride) * ld;
  int kb = k0 + ((lane >> 4) << 3);
  #pragma unroll
  for (int e = 0; e < 16; ++e) a.u[e] = r[kb + e + ((e >> 3) << 3)];
  return a.v;
}

// B fragment 32x16 bf16 from weight stored (N rows, K cols) row-major:
// lane n = n0+(lane&15); elems 0..7 -> K+0..7, elems 8..15 -> K+16..23; lanes 16-31 add +8 to K.
__device__ __forceinline__ bf16x16 load_b(const unsigned short* p, int ld, int n0,
                                          int k0, int lane) {
  ABfrag b;
  const unsigned short* r = p + (size_t)(n0 + (lane & 15)) * ld + k0 + ((lane >> 4) << 3);
  #pragma unroll
  for (int e = 0; e < 16; ++e) b.u[e] = r[e + ((e >> 3) << 3)];
  return b.v;
}

__device__ __forceinline__ f32x8 wmma_bf16(bf16x16 a, bf16x16 b, f32x8 c) {
  return __builtin_amdgcn_wmma_f32_16x16x32_bf16(false, a, false, b, (short)0, c, false, false);
}

#define F32X8_ZERO {0.f, 0.f, 0.f, 0.f, 0.f, 0.f, 0.f, 0.f}

// ---------------- conversion kernels ----------------
__global__ void k_cvt(const float* __restrict__ in, unsigned short* __restrict__ out, int n) {
  for (int i = blockIdx.x * blockDim.x + threadIdx.x; i < n; i += gridDim.x * blockDim.x)
    out[i] = f2bf(in[i]);
}

__global__ void k_cvt_table(const float* __restrict__ in, unsigned short* __restrict__ out) {
  const int n = NUM_ITEMS * D;
  for (int i = blockIdx.x * blockDim.x + threadIdx.x; i < n; i += gridDim.x * blockDim.x)
    out[i] = (i < D) ? (unsigned short)0 : f2bf(in[i]);   // table.at[0] = 0
}

// ---------------- gather + EMA scan ----------------
__global__ void k_ema(const int* __restrict__ state, const float* __restrict__ table,
                      float* __restrict__ embclick, unsigned short* __restrict__ embclick_bf,
                      float* __restrict__ uprof) {
  int g = blockIdx.x * blockDim.x + threadIdx.x;   // Bb*D threads
  int b = g / D, d = g % D;
  float prof = 0.f;
  for (int s = 0; s < S; ++s) {
    int idx = state[(b * S + s) * (1 + Kslate)];
    float e = (idx == 0) ? 0.f : table[(size_t)idx * D + d];
    embclick[(size_t)(b * S + s) * D + d] = e;
    embclick_bf[(size_t)(b * S + s) * D + d] = f2bf(e);
    if (idx != 0) prof = ALPHA * e + (1.f - ALPHA) * prof;
  }
  uprof[b * D + d] = prof;
}

// ---------------- slate gather (async->LDS) + fc1 WMMA + relu + mean over K ----------------
__global__ __launch_bounds__(256) void k_pool(const int* __restrict__ state,
                                              const unsigned short* __restrict__ tab_bf,
                                              const unsigned short* __restrict__ fc1w_bf,
                                              const float* __restrict__ fc1b,
                                              unsigned short* __restrict__ pooled_bf) {
  __shared__ unsigned short As[16 * Kslate * D];   // 160 rows x 128 bf16, 40 KB
  int tid = threadIdx.x;
  int bs0 = blockIdx.x * 16;
#if HAVE_ASYNC_LDS
  // 160 rows x 256 B = 2560 16-byte chunks, DMA'd straight into LDS (ASYNCcnt path)
  for (int j = tid; j < 16 * Kslate * 16; j += 256) {
    int r = j >> 4, ch = j & 15;
    int bs = bs0 + r / Kslate, k = r % Kslate;
    int idx = state[bs * (1 + Kslate) + 1 + k];
    const unsigned short* g = tab_bf + (size_t)idx * D + ch * 8;
    __builtin_amdgcn_global_load_async_to_lds_b128(
        (v4i_gp)g, (v4i_lp)(As + j * 8), 0, 0);
  }
  __builtin_amdgcn_s_wait_asynccnt(0);
#else
  for (int i = tid; i < 16 * Kslate * D; i += 256) {
    int r = i / D, d = i % D;
    int bs = bs0 + r / Kslate, k = r % Kslate;
    int idx = state[bs * (1 + Kslate) + 1 + k];
    As[i] = tab_bf[(size_t)idx * D + d];           // row 0 is pre-zeroed
  }
#endif
  __syncthreads();
  int wave = tid >> 5, lane = tid & 31;
  int n0 = blockIdx.y * 128 + wave * 16;
  bf16x16 bfr[4];
  #pragma unroll
  for (int kk = 0; kk < 4; ++kk) bfr[kk] = load_b(fc1w_bf, D, n0, kk * 32, lane);
  f32x8 pool = F32X8_ZERO;
  for (int k = 0; k < Kslate; ++k) {
    f32x8 c = F32X8_ZERO;
    #pragma unroll
    for (int kk = 0; kk < 4; ++kk)
      c = wmma_bf16(load_a(As, D, k, Kslate, kk * 32, lane), bfr[kk], c);
    int n = n0 + (lane & 15);
    #pragma unroll
    for (int i = 0; i < 8; ++i) pool[i] += fmaxf(c[i] + fc1b[n], 0.f);
  }
  int n = n0 + (lane & 15);
  #pragma unroll
  for (int i = 0; i < 8; ++i) {
    int m = i + ((lane >> 4) << 3);
    pooled_bf[(size_t)(bs0 + m) * H + n] = f2bf(pool[i] * (1.0f / Kslate));
  }
}

// ---------------- fc2 WMMA: ds = relu(pooled @ fc2^T) ----------------
__global__ __launch_bounds__(256) void k_fc2(const unsigned short* __restrict__ pooled_bf,
                                             const unsigned short* __restrict__ fc2w_bf,
                                             const float* __restrict__ fc2b,
                                             float* __restrict__ ds,
                                             unsigned short* __restrict__ ds_bf) {
  int tid = threadIdx.x, wave = tid >> 5, lane = tid & 31;
  int m0 = blockIdx.x * 64;
  int mi = wave >> 1;                 // constant M-tile per wave: hoist A fragments
  int row0 = m0 + mi * 16;
  bf16x16 afr[8];
  #pragma unroll
  for (int kk = 0; kk < 8; ++kk) afr[kk] = load_a(pooled_bf, H, row0, 1, kk * 32, lane);
  for (int t = 0; t < 8; ++t) {
    int n0 = ((wave & 1) * 8 + t) * 16;
    f32x8 c = F32X8_ZERO;
    #pragma unroll
    for (int kk = 0; kk < 8; ++kk)
      c = wmma_bf16(afr[kk], load_b(fc2w_bf, H, n0, kk * 32, lane), c);
    int n = n0 + (lane & 15);
    #pragma unroll
    for (int i = 0; i < 8; ++i) {
      int m = row0 + i + ((lane >> 4) << 3);
      float v = fmaxf(c[i] + fc2b[n], 0.f);
      ds[(size_t)m * H + n] = v;
      ds_bf[(size_t)m * H + n] = f2bf(v);
    }
  }
}

// ---------------- gi = [emb_click, ds] @ wih^T + bih ----------------
__global__ __launch_bounds__(256) void k_gi(const unsigned short* __restrict__ embclick_bf,
                                            const unsigned short* __restrict__ ds_bf,
                                            const unsigned short* __restrict__ wih_bf,
                                            const float* __restrict__ bih,
                                            float* __restrict__ gi) {
  int tid = threadIdx.x, wave = tid >> 5, lane = tid & 31;
  int m0 = blockIdx.x * 64;
  int nblk = blockIdx.y * 64;
  int mi = wave >> 1;                 // constant M-tile per wave
  int row0 = m0 + mi * 16;
  bf16x16 afr[12];                    // K = 384 = 12 fragments
  #pragma unroll
  for (int kk = 0; kk < 4; ++kk)  afr[kk] = load_a(embclick_bf, D, row0, 1, kk * 32, lane);
  #pragma unroll
  for (int kk = 4; kk < 12; ++kk) afr[kk] = load_a(ds_bf, H, row0, 1, (kk - 4) * 32, lane);
  for (int t = 0; t < 2; ++t) {
    int ni = (wave * 2 + t) & 3;
    int n0 = nblk + ni * 16;
    f32x8 c = F32X8_ZERO;
    #pragma unroll
    for (int kk = 0; kk < 12; ++kk)
      c = wmma_bf16(afr[kk], load_b(wih_bf, D + H, n0, kk * 32, lane), c);
    int n = n0 + (lane & 15);
    #pragma unroll
    for (int i = 0; i < 8; ++i) {
      int m = row0 + i + ((lane >> 4) << 3);
      gi[(size_t)m * IN_DIM + n] = c[i] + bih[n];
    }
  }
}

// ---------------- persistent GRU: 32 waves, ping-pong bf16 h in LDS ----------------
__global__ __launch_bounds__(1024) void k_gru(const float* __restrict__ gi,
                                              const unsigned short* __restrict__ whh_bf,
                                              const float* __restrict__ bhh,
                                              float* __restrict__ gructx) {
  extern __shared__ char smem[];
  unsigned short* h0 = (unsigned short*)smem;                       // 128x256 bf16
  unsigned short* h1 = (unsigned short*)(smem + (size_t)Bb * H * 2);
  int tid = threadIdx.x, wave = tid >> 5, lane = tid & 31;
  for (int i = tid; i < 2 * Bb * H; i += 1024) ((unsigned short*)smem)[i] = 0;
  __syncthreads();
  int mi = wave >> 2;                 // (wave*4+j)>>4 is constant per wave
  for (int s = 0; s < S; ++s) {
    const unsigned short* hr = (s & 1) ? h1 : h0;
    unsigned short* hw = (s & 1) ? h0 : h1;
    bf16x16 afr[8];                   // shared by all 4 j-tiles x 3 gates
    #pragma unroll
    for (int kk = 0; kk < 8; ++kk) afr[kk] = load_a(hr, H, mi * 16, 1, kk * 32, lane);
    #pragma unroll
    for (int j = 0; j < 4; ++j) {
      int ni = (wave * 4 + j) & 15;
      int col = ni * 16 + (lane & 15);
      f32x8 cg[3];
      #pragma unroll
      for (int g = 0; g < 3; ++g) {
        f32x8 c = F32X8_ZERO;
        #pragma unroll
        for (int kk = 0; kk < 8; ++kk)
          c = wmma_bf16(afr[kk], load_b(whh_bf, H, g * H + ni * 16, kk * 32, lane), c);
        cg[g] = c;
      }
      #pragma unroll
      for (int i = 0; i < 8; ++i) {
        int bb = mi * 16 + i + ((lane >> 4) << 3);
        const float* gr = gi + (size_t)(bb * S + s) * IN_DIM;
        float ir = gr[col], iz = gr[H + col], inn = gr[2 * H + col];
        float r = sigmf(ir + cg[0][i] + bhh[col]);
        float z = sigmf(iz + cg[1][i] + bhh[H + col]);
        float nn = tanhf(inn + r * (cg[2][i] + bhh[2 * H + col]));
        float hold = bf2f(hr[bb * H + col]);
        hw[bb * H + col] = f2bf((1.f - z) * nn + z * hold);
      }
    }
    __syncthreads();                  // single barrier per step (ping-pong buffers)
  }
  const unsigned short* hf = (S & 1) ? h1 : h0;
  for (int i = tid; i < Bb * H; i += 1024) gructx[i] = bf2f(hf[i]);
}

// ---------------- slate_context = mean_s ds ----------------
__global__ void k_sctx(const float* __restrict__ ds, float* __restrict__ sctx) {
  int g = blockIdx.x * blockDim.x + threadIdx.x;   // Bb*H
  int b = g / H, n = g % H;
  float a = 0.f;
  for (int s = 0; s < S; ++s) a += ds[(size_t)(b * S + s) * H + n];
  sctx[g] = a * (1.0f / S);
}

// ---------------- cosine sim + shifted logits p1/p2 ----------------
__global__ __launch_bounds__(256) void k_sim(const float* __restrict__ uprof,
                                             const float* __restrict__ items,
                                             const float* __restrict__ s1p,
                                             const float* __restrict__ s2p,
                                             float* __restrict__ simw,
                                             float* __restrict__ out) {
  __shared__ float up[D];
  __shared__ float red[256];
  int b = blockIdx.x, tid = threadIdx.x;
  if (tid < D) up[tid] = uprof[b * D + tid];
  __syncthreads();
  red[tid] = (tid < D) ? up[tid] * up[tid] : 0.f;
  __syncthreads();
  for (int o = 128; o > 0; o >>= 1) { if (tid < o) red[tid] += red[tid + o]; __syncthreads(); }
  float unrm = fmaxf(sqrtf(red[0]), 1e-12f);
  __syncthreads();
  float lmax = -1e30f;
  for (int c = tid; c < C; c += 256) {
    const float* ie = items + (size_t)(b * C + c) * D;
    float dot = 0.f, nn = 0.f;
    for (int d = 0; d < D; ++d) { float v = ie[d]; dot += up[d] * v; nn += v * v; }
    float sim = dot / (unrm * fmaxf(sqrtf(nn), 1e-12f));
    simw[b * C + c] = sim;
    lmax = fmaxf(lmax, sim);
  }
  red[tid] = lmax;
  __syncthreads();
  for (int o = 128; o > 0; o >>= 1) { if (tid < o) red[tid] = fmaxf(red[tid], red[tid + o]); __syncthreads(); }
  float smax = red[0];
  float s1 = s1p[0], s2 = s2p[0];
  for (int c = tid; c < C; c += 256) {
    float sim = simw[b * C + c];
    out[2 * BC + b * C + c] = s1 * (sim - smax);   // p1
    out[4 * BC + b * C + c] = s2 * (sim - smax);   // p2
  }
}

// ---------------- fold constant xu columns into per-b bias vector ----------------
__global__ void k_bias(const float* __restrict__ uprof, const float* __restrict__ sctx,
                       const float* __restrict__ gctx, const float* __restrict__ w,
                       const float* __restrict__ bvec, float* __restrict__ bias) {
  int g = blockIdx.x * blockDim.x + threadIdx.x;   // Bb*H
  int b = g / H, n = g % H;
  const float* wr = w + (size_t)n * IN_DIM;
  float a = bvec[n];
  for (int j = 0; j < D; ++j) a += uprof[b * D + j] * wr[j];
  for (int j = 0; j < H; ++j) a += sctx[b * H + j] * wr[D + j];
  for (int j = 0; j < H; ++j) a += gctx[b * H + j] * wr[D + H + j];
  bias[g] = a;
}

// ---------------- fused twin towers: 2x (WMMA l1 -> WMMA l2 -> head) ----------------
__global__ __launch_bounds__(256) void k_tower(const float* __restrict__ items,
    const unsigned short* __restrict__ l1w_bf, const float* __restrict__ bias1,
    const unsigned short* __restrict__ l2w_bf, const float* __restrict__ l2b,
    const float* __restrict__ v1w, const float* __restrict__ v1b,
    const unsigned short* __restrict__ l4w_bf, const float* __restrict__ bias2,
    const unsigned short* __restrict__ l5w_bf, const float* __restrict__ l5b,
    const float* __restrict__ v2w, const float* __restrict__ v2b,
    float* __restrict__ out) {
  __shared__ unsigned short As[64 * D];   // item tile, 16 KB
  __shared__ unsigned short Xs[64 * H];   // activation tile, 32 KB
  int b = blockIdx.x, c0 = blockIdx.y * 64;
  int tid = threadIdx.x, wave = tid >> 5, lane = tid & 31;
  for (int i = tid; i < 64 * D; i += 256) {
    int r = i / D, d = i % D;
    int c = c0 + r;
    As[i] = (c < C) ? f2bf(items[(size_t)(b * C + c) * D + d]) : (unsigned short)0;
  }
  __syncthreads();
  int mi = wave >> 1;                 // constant M-tile per wave
  for (int tw = 0; tw < 2; ++tw) {
    const unsigned short* w1 = tw ? l4w_bf : l1w_bf;
    const float* bs1 = tw ? bias2 : bias1;
    const unsigned short* w2 = tw ? l5w_bf : l2w_bf;
    const float* b2 = tw ? l5b : l2b;
    const float* vw = tw ? v2w : v1w;
    const float* vb = tw ? v2b : v1b;
    // layer 1: only K columns 640..767 of l1_w remain (rest folded into bias)
    {
      bf16x16 afr[4];
      #pragma unroll
      for (int kk = 0; kk < 4; ++kk) afr[kk] = load_a(As, D, mi * 16, 1, kk * 32, lane);
      for (int t = 0; t < 8; ++t) {
        int n0 = ((wave & 1) * 8 + t) * 16;
        f32x8 c = F32X8_ZERO;
        #pragma unroll
        for (int kk = 0; kk < 4; ++kk)
          c = wmma_bf16(afr[kk], load_b(w1, IN_DIM, n0, (IN_DIM - D) + kk * 32, lane), c);
        int n = n0 + (lane & 15);
        #pragma unroll
        for (int i = 0; i < 8; ++i) {
          int m = mi * 16 + i + ((lane >> 4) << 3);
          Xs[m * H + n] = f2bf(fmaxf(c[i] + bs1[b * H + n], 0.f));
        }
      }
    }
    __syncthreads();
    // layer 2: K = 256, results staged in regs, then written back after sync
    float res[8][8];
    for (int t = 0; t < 8; ++t) {
      int n0 = ((wave & 1) * 8 + t) * 16;
      f32x8 c = F32X8_ZERO;
      #pragma unroll
      for (int kk = 0; kk < 8; ++kk)
        c = wmma_bf16(load_a(Xs, H, mi * 16, 1, kk * 32, lane),
                      load_b(w2, H, n0, kk * 32, lane), c);
      int n = n0 + (lane & 15);
      #pragma unroll
      for (int i = 0; i < 8; ++i) res[t][i] = fmaxf(c[i] + b2[n], 0.f);
    }
    __syncthreads();
    for (int t = 0; t < 8; ++t) {
      int n = ((wave & 1) * 8 + t) * 16 + (lane & 15);
      #pragma unroll
      for (int i = 0; i < 8; ++i) {
        int m = mi * 16 + i + ((lane >> 4) << 3);
        Xs[m * H + n] = f2bf(res[t][i]);
      }
    }
    __syncthreads();
    // value head + q = v * sigmoid(p)
    if (tid < 64) {
      int c = c0 + tid;
      if (c < C) {
        float a = vb[0];
        for (int n = 0; n < H; ++n) a += bf2f(Xs[tid * H + n]) * vw[n];
        float p = out[(tw ? 4 : 2) * BC + b * C + c];
        out[(tw ? 5 : 3) * BC + b * C + c] = a;             // v1 / v2
        out[(tw ? 1 : 0) * BC + b * C + c] = a * sigmf(p);  // q1 / q2
      }
    }
    __syncthreads();
  }
}

extern "C" void kernel_launch(void* const* d_in, const int* in_sizes, int n_in,
                              void* d_out, int out_size, void* d_ws, size_t ws_size,
                              hipStream_t stream) {
  const int*   state = (const int*)d_in[0];
  const float* items = (const float*)d_in[1];
  const float* table = (const float*)d_in[2];
  const float* fc1w = (const float*)d_in[3];
  const float* fc1b = (const float*)d_in[4];
  const float* fc2w = (const float*)d_in[5];
  const float* fc2b = (const float*)d_in[6];
  const float* wih  = (const float*)d_in[7];
  const float* whh  = (const float*)d_in[8];
  const float* bih  = (const float*)d_in[9];
  const float* bhh  = (const float*)d_in[10];
  const float* l1w = (const float*)d_in[11];
  const float* l1b = (const float*)d_in[12];
  const float* l2w = (const float*)d_in[13];
  const float* l2b = (const float*)d_in[14];
  const float* v1w = (const float*)d_in[15];
  const float* v1b = (const float*)d_in[16];
  const float* p1s = (const float*)d_in[17];
  const float* l4w = (const float*)d_in[18];
  const float* l4b = (const float*)d_in[19];
  const float* l5w = (const float*)d_in[20];
  const float* l5b = (const float*)d_in[21];
  const float* v2w = (const float*)d_in[22];
  const float* v2b = (const float*)d_in[23];
  const float* p2s = (const float*)d_in[24];
  float* out = (float*)d_out;

  char* ws = (char*)d_ws;
  size_t off = 0;
  auto alloc = [&](size_t bytes) -> char* {
    char* p = ws + off;
    off += (bytes + 255) & ~(size_t)255;
    return p;
  };
  unsigned short* tab_bf      = (unsigned short*)alloc((size_t)NUM_ITEMS * D * 2);
  unsigned short* fc1w_bf     = (unsigned short*)alloc((size_t)H * D * 2);
  unsigned short* fc2w_bf     = (unsigned short*)alloc((size_t)H * H * 2);
  unsigned short* wih_bf      = (unsigned short*)alloc((size_t)3 * H * (D + H) * 2);
  unsigned short* whh_bf      = (unsigned short*)alloc((size_t)3 * H * H * 2);
  unsigned short* l1w_bf      = (unsigned short*)alloc((size_t)H * IN_DIM * 2);
  unsigned short* l2w_bf      = (unsigned short*)alloc((size_t)H * H * 2);
  unsigned short* l4w_bf      = (unsigned short*)alloc((size_t)H * IN_DIM * 2);
  unsigned short* l5w_bf      = (unsigned short*)alloc((size_t)H * H * 2);
  float*          embclick    = (float*)alloc((size_t)BS * D * 4);
  unsigned short* embclick_bf = (unsigned short*)alloc((size_t)BS * D * 2);
  float*          uprof       = (float*)alloc((size_t)Bb * D * 4);
  unsigned short* pooled_bf   = (unsigned short*)alloc((size_t)BS * H * 2);
  float*          dsw         = (float*)alloc((size_t)BS * H * 4);
  unsigned short* ds_bf       = (unsigned short*)alloc((size_t)BS * H * 2);
  float*          gi          = (float*)alloc((size_t)BS * IN_DIM * 4);
  float*          gructx      = (float*)alloc((size_t)Bb * H * 4);
  float*          sctx        = (float*)alloc((size_t)Bb * H * 4);
  float*          simw        = (float*)alloc((size_t)BC * 4);
  float*          bias1       = (float*)alloc((size_t)Bb * H * 4);
  float*          bias2       = (float*)alloc((size_t)Bb * H * 4);
  (void)in_sizes; (void)n_in; (void)out_size; (void)ws_size;

  // precision conversion (one pass; ~80 MB traffic, negligible at 23.3 TB/s)
  k_cvt_table<<<2048, 256, 0, stream>>>(table, tab_bf);
  k_cvt<<<128, 256, 0, stream>>>(fc1w, fc1w_bf, H * D);
  k_cvt<<<256, 256, 0, stream>>>(fc2w, fc2w_bf, H * H);
  k_cvt<<<512, 256, 0, stream>>>(wih, wih_bf, 3 * H * (D + H));
  k_cvt<<<512, 256, 0, stream>>>(whh, whh_bf, 3 * H * H);
  k_cvt<<<512, 256, 0, stream>>>(l1w, l1w_bf, H * IN_DIM);
  k_cvt<<<256, 256, 0, stream>>>(l2w, l2w_bf, H * H);
  k_cvt<<<512, 256, 0, stream>>>(l4w, l4w_bf, H * IN_DIM);
  k_cvt<<<256, 256, 0, stream>>>(l5w, l5w_bf, H * H);

  k_ema<<<(Bb * D) / 128, 128, 0, stream>>>(state, table, embclick, embclick_bf, uprof);
  k_pool<<<dim3(BS / 16, 2), 256, 0, stream>>>(state, tab_bf, fc1w_bf, fc1b, pooled_bf);
  k_fc2<<<BS / 64, 256, 0, stream>>>(pooled_bf, fc2w_bf, fc2b, dsw, ds_bf);
  k_gi<<<dim3(BS / 64, IN_DIM / 64), 256, 0, stream>>>(embclick_bf, ds_bf, wih_bf, bih, gi);
  k_gru<<<1, 1024, (size_t)Bb * H * 2 * 2, stream>>>(gi, whh_bf, bhh, gructx);
  k_sctx<<<(Bb * H) / 256, 256, 0, stream>>>(dsw, sctx);
  k_sim<<<Bb, 256, 0, stream>>>(uprof, items, p1s, p2s, simw, out);
  k_bias<<<(Bb * H) / 256, 256, 0, stream>>>(uprof, sctx, gructx, l1w, l1b, bias1);
  k_bias<<<(Bb * H) / 256, 256, 0, stream>>>(uprof, sctx, gructx, l4w, l4b, bias2);
  k_tower<<<dim3(Bb, (C + 63) / 64), 256, 0, stream>>>(items, l1w_bf, bias1, l2w_bf, l2b,
                                                       v1w, v1b, l4w_bf, bias2, l5w_bf, l5b,
                                                       v2w, v2b, out);
}
